// Block_7052336300645
// MI455X (gfx1250) — matrix operand order repaired and checked by reading, not compiled, not run on hardware
//
#include <hip/hip_runtime.h>
#include <hip/hip_bf16.h>
#include <cstdint>
#include <cstddef>

// ---------------------------------------------------------------------------
// Types for CDNA5 WMMA (gfx1250, wave32)
// ---------------------------------------------------------------------------
typedef __attribute__((ext_vector_type(16))) __bf16 v16bf;
typedef __attribute__((ext_vector_type(8)))  float  v8f;

union ABFrag {
    uint4 u[2];   // 2 x 16B = 32B
    v16bf v;      // 16 x bf16 = 32B
};

// ---------------------------------------------------------------------------
// f32 -> bf16 conversion (weights)
// ---------------------------------------------------------------------------
__global__ void convert_f32_bf16(const float* __restrict__ in,
                                 __bf16* __restrict__ out, long n) {
    long i = (long)blockIdx.x * blockDim.x + threadIdx.x;
    if (i < n) out[i] = (__bf16)in[i];
}

// ---------------------------------------------------------------------------
// LayerNorm + time-shift token mixing.
// For row (b,t): xn = LN(x[b,t]); xx = (t==0 ? shift[b] : LN(x[b,t-1])).
// out_j = bf16( xn*mix_j + xx*(1-mix_j) ).  Optionally writes xn of the last
// timestep to last_out (the new shift state).
// ---------------------------------------------------------------------------
__global__ __launch_bounds__(256)
void ln_mix_kernel(const float* __restrict__ x, const float* __restrict__ shift,
                   const float* __restrict__ g, const float* __restrict__ be,
                   const float* __restrict__ mk, const float* __restrict__ mv,
                   const float* __restrict__ mr,
                   __bf16* __restrict__ ok, __bf16* __restrict__ ov,
                   __bf16* __restrict__ orr,
                   float* __restrict__ last_out,
                   int Bn, int Tn, int Cn)
{
    __shared__ float red[256];
    const int tid = threadIdx.x;
    const int row = blockIdx.x;
    const int b = row / Tn;
    const int t = row - b * Tn;

    const float* xc = x + (size_t)row * Cn;
    const bool prevLN = (t != 0);
    const float* xp = prevLN ? (x + (size_t)(row - 1) * Cn)
                             : (shift + (size_t)b * Cn);

    float s0 = 0.f, s1 = 0.f, s2 = 0.f, s3 = 0.f;
    for (int i = tid; i < Cn; i += 256) {
        float vc = xc[i];
        s0 += vc; s1 += vc * vc;
        if (prevLN) { float vp = xp[i]; s2 += vp; s3 += vp * vp; }
    }
    auto blockReduce = [&](float v) -> float {
        red[tid] = v; __syncthreads();
        for (int s = 128; s > 0; s >>= 1) {
            if (tid < s) red[tid] += red[tid + s];
            __syncthreads();
        }
        float r = red[0]; __syncthreads();
        return r;
    };
    const float invC = 1.0f / (float)Cn;
    float S0 = blockReduce(s0);
    float S1 = blockReduce(s1);
    float mu = S0 * invC;
    float var = S1 * invC - mu * mu;
    float rstd = rsqrtf(var + 1e-5f);

    float muP = 0.f, rstdP = 0.f;
    if (prevLN) {
        float S2 = blockReduce(s2);
        float S3 = blockReduce(s3);
        muP = S2 * invC;
        float varP = S3 * invC - muP * muP;
        rstdP = rsqrtf(varP + 1e-5f);
    }

    const bool lastT = (t == Tn - 1) && (last_out != nullptr);
    for (int i = tid; i < Cn; i += 256) {
        float gi = g[i], bi = be[i];
        float xn = (xc[i] - mu) * rstd * gi + bi;
        float xxv = prevLN ? ((xp[i] - muP) * rstdP * gi + bi) : xp[i];
        size_t o = (size_t)row * Cn + i;
        {
            float m = mk[i];
            ok[o] = (__bf16)(xn * m + xxv * (1.0f - m));
        }
        if (ov) {
            float m = mv[i];
            ov[o] = (__bf16)(xn * m + xxv * (1.0f - m));
        }
        if (orr) {
            float m = mr[i];
            orr[o] = (__bf16)(xn * m + xxv * (1.0f - m));
        }
        if (lastT) last_out[(size_t)b * Cn + i] = xn;
    }
}

// ---------------------------------------------------------------------------
// WKV recurrence: one thread per (b, a) channel, sequential over T.
// ---------------------------------------------------------------------------
__global__ __launch_bounds__(256)
void wkv_scan_kernel(const float* __restrict__ td, const float* __restrict__ tf,
                     const float* __restrict__ k, const float* __restrict__ v,
                     const float* __restrict__ r,
                     const float* __restrict__ state_in,
                     float* __restrict__ state_out,
                     __bf16* __restrict__ sy,
                     int Bn, int Tn, int An)
{
    int idx = blockIdx.x * blockDim.x + threadIdx.x;
    if (idx >= Bn * An) return;
    int b = idx / An;
    int a = idx - b * An;

    const float w = -__expf(td[a]);
    const float u = tf[a];

    size_t base = (size_t)b * Tn * An + a;
    const float* kp = k + base;
    const float* vp = v + base;
    const float* rp = r + base;
    __bf16* syp = sy + base;

    size_t sidx = ((size_t)b * An + a) * 3;
    float aa = state_in[sidx + 0];
    float bb = state_in[sidx + 1];
    float pp = state_in[sidx + 2];

    for (int t = 0; t < Tn; ++t) {
        size_t o = (size_t)t * An;
        float kt = kp[o];
        float vt = vp[o];
        float rt = rp[o];

        float ww = u + kt;
        float p  = fmaxf(pp, ww);
        float e1 = __expf(pp - p);
        float e2 = __expf(ww - p);
        float y  = (e1 * aa + e2 * vt) / (e1 * bb + e2);

        float sr = 1.0f / (1.0f + __expf(-rt));
        syp[o] = (__bf16)(sr * y);

        float ww2 = pp + w;
        float p2  = fmaxf(ww2, kt);
        e1 = __expf(ww2 - p2);
        e2 = __expf(kt - p2);
        aa = e1 * aa + e2 * vt;
        bb = e1 * bb + e2;
        pp = p2;
    }
    state_out[sidx + 0] = aa;
    state_out[sidx + 1] = bb;
    state_out[sidx + 2] = pp;
}

// ---------------------------------------------------------------------------
// bf16 WMMA GEMM:  out[M,N] = act[M,K] * W[N,K]^T   (f32 accumulate)
// Block tile 128x128, K-step 64, 8 waves (2x4), wave tile 64x32:
// 16 x v_wmma_f32_16x16x32_bf16 per iteration.
// Global->LDS staging uses gfx1250 async copies (ASYNCcnt), triple-buffered
// so only ONE workgroup barrier per iteration is needed:
//   - issue at iter t writes buf[(t+1)%3], last read at iter t-1... t-2;
//     the barrier at iter t-1 orders those reads before this write.
// Epilogues:
//   0: outF = acc
//   1: outF = res + acc                      (attention residual)
//   2: outB = bf16(relu(acc)^2)              (FFN k path)
//   3: outF = res + sigmoid(sig)*acc         (final FFN merge, in-place)
// Requires M%128==0, N%128==0, K%64==0 (true for all shapes here).
// ---------------------------------------------------------------------------
#define BM 128
#define BN 128
#define BKS 64
#define LDSP 80   // padded LDS row stride in bf16 elems (160B, 16B multiple)

__global__ __launch_bounds__(256)
void wmma_gemm_bf16(const __bf16* __restrict__ Am, const __bf16* __restrict__ Wm,
                    float* __restrict__ outF, __bf16* __restrict__ outB,
                    const float* __restrict__ res, const float* __restrict__ sig,
                    int M, int N, int K, int epi)
{
    __shared__ __align__(16) __bf16 lA[3][BM][LDSP];
    __shared__ __align__(16) __bf16 lB[3][BN][LDSP];

    const int tid  = threadIdx.x;
    const int lane = tid & 31;
    const int wave = tid >> 5;
    const int wm = wave >> 2;          // 0..1  -> 64 rows
    const int wn = wave & 3;           // 0..3  -> 32 cols
    const int bm0 = blockIdx.y * BM;
    const int bn0 = blockIdx.x * BN;

    const v8f vzero = {0.f, 0.f, 0.f, 0.f, 0.f, 0.f, 0.f, 0.f};
    v8f acc[4][2];
#pragma unroll
    for (int mi = 0; mi < 4; ++mi)
#pragma unroll
        for (int ni = 0; ni < 2; ++ni) acc[mi][ni] = vzero;

    const int khalf = lane >> 4;       // 0 or 1
    const int lrow  = lane & 15;

    // Issue async global->LDS copies for one 128x64 A tile + B tile.
    // 1024 x 16B chunks per matrix => 4 chunks/thread/matrix;
    // per wave: 8 x global_load_async_to_lds_b128 (ASYNCcnt += 8).
    auto stage_async = [&](int kk, int pb) {
#pragma unroll
        for (int p = 0; p < 4; ++p) {
            int chunk = p * 256 + tid;
            int rr = chunk >> 3;            // 0..127
            int ch = (chunk & 7) * 8;       // bf16 elem offset within row
            unsigned long long ga =
                (unsigned long long)(Am + (size_t)(bm0 + rr) * K + kk + ch);
            unsigned la = (unsigned)(uintptr_t)&lA[pb][rr][ch];
            asm volatile("global_load_async_to_lds_b128 %0, %1, off"
                         :: "v"(la), "v"(ga) : "memory");
            unsigned long long gb =
                (unsigned long long)(Wm + (size_t)(bn0 + rr) * K + kk + ch);
            unsigned lb = (unsigned)(uintptr_t)&lB[pb][rr][ch];
            asm volatile("global_load_async_to_lds_b128 %0, %1, off"
                         :: "v"(lb), "v"(gb) : "memory");
        }
    };

    const int nt = K / BKS;
    stage_async(0, 0);

    int cur = 0, nxt = 1;
    for (int t = 0; t < nt; ++t) {
        if (t + 1 < nt) {
            // Fetch next tile into the buffer read two iterations ago, then
            // wait only for the current tile (ASYNCcnt retires in order:
            // <=8 outstanding means tile t is complete).
            stage_async((t + 1) * BKS, nxt);
            asm volatile("s_wait_asynccnt 0x8" ::: "memory");
        } else {
            asm volatile("s_wait_asynccnt 0x0" ::: "memory");
        }
        __syncthreads();   // all waves' async tiles visible in LDS

        // Two K=32 sub-steps per 64-wide tile: 16 WMMAs per barrier.
#pragma unroll
        for (int s = 0; s < 2; ++s) {
            const int kb = s * 32;
            // B fragments: lane<16 holds K 0..15, lane>=16 holds K 16..31
            ABFrag bfr[2];
#pragma unroll
            for (int ni = 0; ni < 2; ++ni) {
                int rr = wn * 32 + ni * 16 + lrow;    // W row == B column
                bfr[ni].u[0] = *(const uint4*)&lB[cur][rr][kb + khalf * 16];
                bfr[ni].u[1] = *(const uint4*)&lB[cur][rr][kb + khalf * 16 + 8];
            }
            // A fragments: lane<16 holds K 0..15, lane>=16 holds K 16..31
#pragma unroll
            for (int mi = 0; mi < 4; ++mi) {
                ABFrag af;
                int rr = wm * 64 + mi * 16 + lrow;
                af.u[0] = *(const uint4*)&lA[cur][rr][kb + khalf * 16];
                af.u[1] = *(const uint4*)&lA[cur][rr][kb + khalf * 16 + 8];
#pragma unroll
                for (int ni = 0; ni < 2; ++ni) {
                    acc[mi][ni] = __builtin_amdgcn_wmma_f32_16x16x32_bf16(
                        false, af.v, false, bfr[ni].v, (short)0, acc[mi][ni],
                        false, false);
                }
            }
        }
        // rotate triple buffer
        cur = nxt;
        nxt = nxt + 1;
        if (nxt == 3) nxt = 0;
    }

    // ---- epilogue: VGPR r holds M=r (lane<16) or M=r+8 (lane>=16), N=lane%16
    const int lhi = (lane >> 4) << 3;
#pragma unroll
    for (int mi = 0; mi < 4; ++mi) {
#pragma unroll
        for (int ni = 0; ni < 2; ++ni) {
            int mbase = bm0 + wm * 64 + mi * 16 + lhi;
            int nn = bn0 + wn * 32 + ni * 16 + lrow;
#pragma unroll
            for (int r2 = 0; r2 < 8; ++r2) {
                size_t idx = (size_t)(mbase + r2) * N + nn;
                float a = acc[mi][ni][r2];
                if (epi == 0) {
                    outF[idx] = a;
                } else if (epi == 1) {
                    outF[idx] = res[idx] + a;
                } else if (epi == 2) {
                    float tv = fmaxf(a, 0.0f);
                    outB[idx] = (__bf16)(tv * tv);
                } else {
                    float sg = 1.0f / (1.0f + __expf(-sig[idx]));
                    outF[idx] = res[idx] + sg * a;
                }
            }
        }
    }
}

// ---------------------------------------------------------------------------
// Host-side orchestration
// ---------------------------------------------------------------------------
extern "C" void kernel_launch(void* const* d_in, const int* in_sizes, int n_in,
                              void* d_out, int out_size, void* d_ws, size_t ws_size,
                              hipStream_t stream) {
    // Inputs (setup_inputs order)
    const float* x        = (const float*)d_in[0];
    const float* tm_shift = (const float*)d_in[1];
    const float* wkv_st   = (const float*)d_in[2];
    const float* cm_shift = (const float*)d_in[3];
    const float* ln1_g = (const float*)d_in[4];
    const float* ln1_b = (const float*)d_in[5];
    const float* ln2_g = (const float*)d_in[6];
    const float* ln2_b = (const float*)d_in[7];
    const float* tdec  = (const float*)d_in[8];
    const float* tfst  = (const float*)d_in[9];
    const float* amk = (const float*)d_in[10];
    const float* amv = (const float*)d_in[11];
    const float* amr = (const float*)d_in[12];
    const float* Wk = (const float*)d_in[13];
    const float* Wv = (const float*)d_in[14];
    const float* Wr = (const float*)d_in[15];
    const float* Wo = (const float*)d_in[16];
    const float* fmk = (const float*)d_in[17];
    const float* fmr = (const float*)d_in[18];
    const float* Wkf = (const float*)d_in[19];
    const float* Wrf = (const float*)d_in[20];
    const float* Wvf = (const float*)d_in[21];

    const int C  = in_sizes[4];                 // 2048
    const int A  = in_sizes[8];                 // 2048
    const int FF = in_sizes[19] / C;            // 8192
    const int B  = in_sizes[1] / C;             // 4
    const int T  = in_sizes[0] / (B * C);       // 4096
    const size_t M = (size_t)B * T;             // 16384

    // Outputs, concatenated flat: x, new_wkv, new_tm_shift, new_cm_shift
    float* xout   = (float*)d_out;
    float* o_wkv  = xout + M * C;
    float* o_tm   = o_wkv + (size_t)B * C * 3;
    float* o_cm   = o_tm + (size_t)B * C;

    // Workspace carve-up
    char* wp = (char*)d_ws;
    auto carve = [&](size_t bytes) -> void* {
        void* p = (void*)wp;
        wp += (bytes + 255) & ~(size_t)255;
        return p;
    };
    __bf16* wk_bf  = (__bf16*)carve((size_t)A * C * 2);
    __bf16* wv_bf  = (__bf16*)carve((size_t)A * C * 2);
    __bf16* wr_bf  = (__bf16*)carve((size_t)A * C * 2);
    __bf16* wo_bf  = (__bf16*)carve((size_t)C * A * 2);
    __bf16* wkf_bf = (__bf16*)carve((size_t)FF * C * 2);
    __bf16* wrf_bf = (__bf16*)carve((size_t)C * C * 2);
    __bf16* wvf_bf = (__bf16*)carve((size_t)C * FF * 2);
    __bf16* bufK   = (__bf16*)carve(M * C * 2);   // xk -> sy
    __bf16* bufV   = (__bf16*)carve(M * C * 2);   // xv -> xkf
    __bf16* bufR   = (__bf16*)carve(M * C * 2);   // xr -> xrf
    __bf16* kf_bf  = (__bf16*)carve(M * FF * 2);  // relu(k_f)^2
    float* kbuf = (float*)carve(M * A * 4);       // k -> rf logits
    float* vbuf = (float*)carve(M * A * 4);       // v
    float* rbuf = (float*)carve(M * A * 4);       // r

    auto cvt = [&](const float* src, __bf16* dst, long n) {
        convert_f32_bf16<<<(unsigned)((n + 255) / 256), 256, 0, stream>>>(src, dst, n);
    };
    cvt(Wk,  wk_bf,  (long)A * C);
    cvt(Wv,  wv_bf,  (long)A * C);
    cvt(Wr,  wr_bf,  (long)A * C);
    cvt(Wo,  wo_bf,  (long)C * A);
    cvt(Wkf, wkf_bf, (long)FF * C);
    cvt(Wrf, wrf_bf, (long)C * C);
    cvt(Wvf, wvf_bf, (long)C * FF);

    // --- attention half ---
    ln_mix_kernel<<<(unsigned)M, 256, 0, stream>>>(
        x, tm_shift, ln1_g, ln1_b, amk, amv, amr,
        bufK, bufV, bufR, o_tm, B, T, C);

    dim3 blk(256);
    dim3 gCA((unsigned)(A / BN), (unsigned)(M / BM));
    wmma_gemm_bf16<<<gCA, blk, 0, stream>>>(bufK, wk_bf, kbuf, nullptr, nullptr, nullptr,
                                            (int)M, A, C, 0);
    wmma_gemm_bf16<<<gCA, blk, 0, stream>>>(bufV, wv_bf, vbuf, nullptr, nullptr, nullptr,
                                            (int)M, A, C, 0);
    wmma_gemm_bf16<<<gCA, blk, 0, stream>>>(bufR, wr_bf, rbuf, nullptr, nullptr, nullptr,
                                            (int)M, A, C, 0);

    wkv_scan_kernel<<<(unsigned)((B * A + 255) / 256), 256, 0, stream>>>(
        tdec, tfst, kbuf, vbuf, rbuf, wkv_st, o_wkv, bufK /* sy */, B, T, A);

    dim3 gAC((unsigned)(C / BN), (unsigned)(M / BM));
    wmma_gemm_bf16<<<gAC, blk, 0, stream>>>(bufK /* sy */, wo_bf, xout, nullptr,
                                            x /* residual */, nullptr,
                                            (int)M, C, A, 1);

    // --- FFN half ---
    ln_mix_kernel<<<(unsigned)M, 256, 0, stream>>>(
        xout, cm_shift, ln2_g, ln2_b, fmk, fmr, nullptr,
        bufV /* xkf */, bufR /* xrf */, nullptr, o_cm, B, T, C);

    dim3 gFF((unsigned)(FF / BN), (unsigned)(M / BM));
    wmma_gemm_bf16<<<gFF, blk, 0, stream>>>(bufV /* xkf */, wkf_bf, nullptr, kf_bf,
                                            nullptr, nullptr, (int)M, FF, C, 2);
    wmma_gemm_bf16<<<gAC, blk, 0, stream>>>(bufR /* xrf */, wrf_bf, kbuf /* rf */,
                                            nullptr, nullptr, nullptr,
                                            (int)M, C, C, 0);
    wmma_gemm_bf16<<<gAC, blk, 0, stream>>>(kf_bf, wvf_bf, xout, nullptr,
                                            xout /* residual */, kbuf /* sig */,
                                            (int)M, C, FF, 3);

    (void)n_in; (void)out_size; (void)ws_size;
}
